// SimpleRNN_30305289240810
// MI455X (gfx1250) — matrix operand order repaired and checked by reading, not compiled, not run on hardware
//
#include <hip/hip_runtime.h>
#include <math.h>

// ---------------------------------------------------------------------------
// SimpleRNN on gfx1250 (MI455X), bf16 WMMA path, v3.
//   B=64, T=512, D=1024, H=1024
//   Phase 0: pack U,W (B-frag layout) and x, state (A-frag layout) to bf16.
//   Phase 1: d_out[b,t,:] = x[b,t,:] @ U + b   (bf16 WMMA, occupancy-hidden)
//   Phase 2: persistent grid-synced scan. W lives in LDS for all 512 steps;
//            h is staged to LDS each step with GLOBAL_LOAD_ASYNC_TO_LDS_B128
//            bursts (ASYNCcnt), so the serial critical path sees only LDS
//            latency in the K-loop instead of 32 serialized L2 round trips.
// ---------------------------------------------------------------------------

typedef __attribute__((ext_vector_type(16))) __bf16        v16bf;
typedef __attribute__((ext_vector_type(8)))  float         v8f;
typedef __attribute__((ext_vector_type(8)))  unsigned int  v8u;

union Frag { v16bf bf; v8u u; };

__device__ __forceinline__ unsigned short f2bf(float f) {
  unsigned u = __float_as_uint(f);
  unsigned r = u + 0x7FFFu + ((u >> 16) & 1u);   // round-to-nearest-even
  return (unsigned short)(r >> 16);
}

__device__ __forceinline__ v8f wmma_bf16(v8u a, v8u b, v8f c) {
  Frag fa; fa.u = a;
  Frag fb; fb.u = b;
  return __builtin_amdgcn_wmma_f32_16x16x32_bf16(
      false, fa.bf, false, fb.bf, (short)0, c, false, false);
}

// LDS byte-offset of a pointer into a __shared__ array (generic -> AS3).
__device__ __forceinline__ unsigned lds_off(const void* p) {
  return (unsigned)(uintptr_t)(__attribute__((address_space(3))) const void*)p;
}

// Async copy 16 bytes global -> LDS per lane (tracked by ASYNCcnt).
__device__ __forceinline__ void async_b128(unsigned loff, const void* gaddr) {
  asm volatile("global_load_async_to_lds_b128 %0, %1, off"
               :: "v"(loff), "v"((unsigned long long)(uintptr_t)gaddr)
               : "memory");
}
__device__ __forceinline__ void wait_async0() {
  asm volatile("s_wait_asynccnt 0x0" ::: "memory");
}

// ---------------------------------------------------------------------------
// B-fragment pack: 1024x1024 fp32 row-major (K x N) -> dst[nt][kt][lane][16]
// bf16.  lane L: n = nt*16 + (L&15), kk = kt*32 + (L>>4)*8 + (i>>3)*16 + (i&7), i = 0..15.
// ---------------------------------------------------------------------------
__global__ void pack_bfrag(const float* __restrict__ src,
                           unsigned short* __restrict__ dst) {
  unsigned tid  = blockIdx.x * blockDim.x + threadIdx.x;  // 0 .. 1M-1
  unsigned i    = tid & 15u;
  unsigned lane = (tid >> 4) & 31u;
  unsigned kt   = (tid >> 9) & 31u;
  unsigned nt   = tid >> 14;
  unsigned n    = nt * 16u + (lane & 15u);
  unsigned kk   = kt * 32u + ((lane >> 4) << 3) + ((i >> 3) << 4) + (i & 7u);
  dst[tid] = f2bf(src[kk * 1024u + n]);
}

// ---------------------------------------------------------------------------
// A-fragment pack: rows x 1024 fp32 (M x K) -> dst[mt][kt][lane][8] u32
// (bf16 pairs). lane L: m = mt*16+(L&15),
// pair v: kk = kt*32 + (L>>4)*8 + (v>>2)*16 + (v&3)*2 (+0,+1).
// ---------------------------------------------------------------------------
__global__ void pack_afrag(const float* __restrict__ src,
                           unsigned int* __restrict__ dst) {
  unsigned tid  = blockIdx.x * blockDim.x + threadIdx.x;
  unsigned v    = tid & 7u;
  unsigned lane = (tid >> 3) & 31u;
  unsigned kt   = (tid >> 8) & 31u;
  unsigned mt   = tid >> 13;
  unsigned m    = mt * 16u + (lane & 15u);
  unsigned kk   = kt * 32u + ((lane >> 4) << 3) + ((v >> 2) << 4) + ((v & 3) << 1);
  float2 f = *(const float2*)(src + (size_t)m * 1024u + kk);
  dst[tid] = (unsigned)f2bf(f.x) | ((unsigned)f2bf(f.y) << 16);
}

__global__ void init_barrier(int* cnt, int* gen) {
  if (threadIdx.x == 0) { cnt[0] = 0; gen[0] = 0; }
}

__device__ __forceinline__ Frag load_afrag_cvt(const float* __restrict__ arow,
                                               int kt, int klane8) {
  Frag a;
#pragma unroll
  for (int v = 0; v < 8; ++v) {
    int kk = kt * 32 + klane8 + ((v >> 2) << 4) + ((v & 3) << 1);
    float2 f = *(const float2*)(arow + kk);
    a.u[v] = (unsigned)f2bf(f.x) | ((unsigned)f2bf(f.y) << 16);
  }
  return a;
}

// ---------------------------------------------------------------------------
// Phase 1: xU GEMM (packed-A).  M=32768, N=1024, K=1024.  8 waves/block,
// wave = 16(M) x 64(N) strip.  Latency hidden by occupancy (32768 waves).
// ---------------------------------------------------------------------------
__global__ void xu_gemm_packed(const unsigned int* __restrict__ Xp,
                               const unsigned int* __restrict__ Up,
                               const float* __restrict__ bias,
                               float* __restrict__ out) {
  const int H = 1024;
  int lane   = threadIdx.x & 31;
  int wave   = threadIdx.x >> 5;
  int strip  = blockIdx.x * 8 + wave;      // 0..32767
  int mt     = strip >> 4;
  int ns     = strip & 15;
  int klane8 = (lane >> 4) << 3;
  int nlane  = lane & 15;

  const v8u* Ap = (const v8u*)Xp + ((size_t)mt * 32) * 32 + lane;
  const v8u* Bp[4];
#pragma unroll
  for (int j = 0; j < 4; ++j)
    Bp[j] = (const v8u*)Up + ((size_t)(ns * 4 + j) * 32) * 32 + lane;

  v8f c[4];
#pragma unroll
  for (int j = 0; j < 4; ++j) c[j] = (v8f){0.f,0.f,0.f,0.f,0.f,0.f,0.f,0.f};

  for (int kt = 0; kt < 32; ++kt) {
    v8u a = Ap[kt * 32];
#pragma unroll
    for (int j = 0; j < 4; ++j)
      c[j] = wmma_bf16(a, Bp[j][kt * 32], c[j]);
  }

#pragma unroll
  for (int j = 0; j < 4; ++j) {
    int ncol = (ns * 4 + j) * 16 + nlane;
    float bb = bias[ncol];
#pragma unroll
    for (int r = 0; r < 8; ++r)
      out[(size_t)(mt * 16 + klane8 + r) * H + ncol] = c[j][r] + bb;
  }
}

// Fallback if ws too small for packed x.
__global__ void xu_gemm_convert(const float* __restrict__ x,
                                const unsigned int* __restrict__ Up,
                                const float* __restrict__ bias,
                                float* __restrict__ out) {
  const int H = 1024, D = 1024;
  int lane   = threadIdx.x & 31;
  int wave   = threadIdx.x >> 5;
  int strip  = blockIdx.x * 8 + wave;
  int mt     = strip >> 4;
  int ns     = strip & 15;
  int klane8 = (lane >> 4) << 3;
  int nlane  = lane & 15;
  const float* arow = x + (size_t)(mt * 16 + nlane) * D;

  v8f c[4];
#pragma unroll
  for (int j = 0; j < 4; ++j) c[j] = (v8f){0.f,0.f,0.f,0.f,0.f,0.f,0.f,0.f};

  for (int kt = 0; kt < 32; ++kt) {
    Frag a = load_afrag_cvt(arow, kt, klane8);
#pragma unroll
    for (int j = 0; j < 4; ++j) {
      const v8u* bp = (const v8u*)Up + ((size_t)(ns * 4 + j) * 32 + kt) * 32 + lane;
      c[j] = wmma_bf16(a.u, *bp, c[j]);
    }
  }
#pragma unroll
  for (int j = 0; j < 4; ++j) {
    int ncol = (ns * 4 + j) * 16 + nlane;
    float bb = bias[ncol];
#pragma unroll
    for (int r = 0; r < 8; ++r)
      out[(size_t)(mt * 16 + klane8 + r) * H + ncol] = c[j][r] + bb;
  }
}

// ---------------------------------------------------------------------------
// Grid-wide barrier (generation based, self-resetting).
// ---------------------------------------------------------------------------
__device__ __forceinline__ void grid_barrier(int* cnt, int* gen, int nblk) {
  __syncthreads();
  __threadfence();
  if (threadIdx.x == 0) {
    int g = __hip_atomic_load(gen, __ATOMIC_ACQUIRE, __HIP_MEMORY_SCOPE_AGENT);
    if (__hip_atomic_fetch_add(cnt, 1, __ATOMIC_ACQ_REL,
                               __HIP_MEMORY_SCOPE_AGENT) == nblk - 1) {
      __hip_atomic_store(cnt, 0, __ATOMIC_RELAXED, __HIP_MEMORY_SCOPE_AGENT);
      __hip_atomic_fetch_add(gen, 1, __ATOMIC_ACQ_REL, __HIP_MEMORY_SCOPE_AGENT);
    } else {
      while (__hip_atomic_load(gen, __ATOMIC_ACQUIRE,
                               __HIP_MEMORY_SCOPE_AGENT) == g) {
        __builtin_amdgcn_s_sleep(1);
      }
    }
  }
  __syncthreads();
}

// ---------------------------------------------------------------------------
// Phase 2: recurrent scan.  16 blocks x 8 waves, wave owns 16(M) x 32(N).
// LDS: Wl = block's 4 N-tiles of packed W (128 KB, loaded once, reused 512x);
//      Hl = full packed h (128 KB), refreshed per step via async-to-LDS burst.
// K-loop reads A and B from LDS only (ds_load_b128), WMMA accumulates fp32.
// ---------------------------------------------------------------------------
__global__ void __launch_bounds__(256, 1)
rnn_scan(const unsigned int* __restrict__ Wp,
         float* __restrict__ out,
         const unsigned int* __restrict__ Hp0,
         unsigned int* __restrict__ Hp1_rw,
         int* cnt, int* gen) {
  __shared__ unsigned int Wl[32768];   // 128 KB: [ntl][kt][lane][8]
  __shared__ unsigned int Hl[32768];   // 128 KB: [mt ][kt][lane][8]

  const int T = 512, H = 1024, B = 64;
  int tid    = threadIdx.x;
  int lane   = tid & 31;
  int wave   = tid >> 5;
  int s      = blockIdx.x * 8 + wave;      // 0..127
  int mt     = s & 3;
  int ng     = s >> 2;                     // global nt = 2*ng + j
  int klane8 = (lane >> 4) << 3;
  int nlane  = lane & 15;
  int nblk   = gridDim.x;

  const unsigned int* Hbuf[2] = { Hp0, Hp1_rw };

  // ---- Stage this block's W slice (global nt = 4*blk .. 4*blk+3) into LDS.
  {
    const char* wsrc = (const char*)Wp + (size_t)blockIdx.x * 131072u;
    unsigned    wl0  = lds_off(Wl);
#pragma unroll
    for (int i = 0; i < 32; ++i) {
      unsigned off = ((unsigned)tid + (unsigned)i * 256u) << 4;
      async_b128(wl0 + off, wsrc + off);
    }
  }

  // Per-lane LDS fragment pointers (v8u units).
  int ntl0 = (ng & 1) * 2;                     // local N-tile within block
  const v8u* Al  = (const v8u*)Hl + (size_t)(mt * 32) * 32 + lane;
  const v8u* Bl0 = (const v8u*)Wl + (size_t)((ntl0 + 0) * 32) * 32 + lane;
  const v8u* Bl1 = (const v8u*)Wl + (size_t)((ntl0 + 1) * 32) * 32 + lane;

  // Packed-h store indices (lane/tile dependent only).
  unsigned hidx[2][8];
#pragma unroll
  for (int j = 0; j < 2; ++j) {
    int hcol = (ng * 2 + j) * 16 + nlane;
    unsigned kt_ = hcol >> 5, kk = hcol & 31;
    unsigned v_  = ((kk >> 4) << 2) | ((kk >> 1) & 3);
    unsigned rem = kk & 1;
#pragma unroll
    for (int r = 0; r < 8; ++r) {
      unsigned lane_ = (unsigned)(klane8 + r) | (((kk >> 3) & 1u) << 4);
      hidx[j][r] = ((((unsigned)mt * 32 + kt_) * 32 + lane_) << 4) + (v_ << 1) + rem;
    }
  }

  wait_async0();
  __syncthreads();                              // Wl ready

  unsigned hl0 = lds_off(Hl);

  for (int t = 0; t < T; ++t) {
    // ---- Burst-copy current h (128 KB) into LDS: 32 async b128 per thread,
    //      no intermediate waits -> ~one L2 round trip total.
    const char* hsrc = (const char*)Hbuf[t & 1];
#pragma unroll
    for (int i = 0; i < 32; ++i) {
      unsigned off = ((unsigned)tid + (unsigned)i * 256u) << 4;
      async_b128(hl0 + off, hsrc + off);
    }

    // ---- Prefetch xu_t while the async copy is in flight.
    float xu[2][8];
#pragma unroll
    for (int j = 0; j < 2; ++j) {
      int hcol = (ng * 2 + j) * 16 + nlane;
#pragma unroll
      for (int r = 0; r < 8; ++r) {
        int brow = mt * 16 + klane8 + r;
        xu[j][r] = out[((size_t)brow * T + t) * H + hcol];
      }
    }

    wait_async0();
    __syncthreads();                            // Hl ready for all waves

    // ---- K loop: pure LDS reads + WMMA.
    v8f c[2];
    c[0] = (v8f){0.f,0.f,0.f,0.f,0.f,0.f,0.f,0.f};
    c[1] = (v8f){0.f,0.f,0.f,0.f,0.f,0.f,0.f,0.f};
    for (int kt = 0; kt < 32; ++kt) {
      v8u a  = Al[kt * 32];
      v8u b0 = Bl0[kt * 32];
      v8u b1 = Bl1[kt * 32];
      c[0] = wmma_bf16(a, b0, c[0]);
      c[1] = wmma_bf16(a, b1, c[1]);
    }

    // ---- Epilogue: tanh, write outs (fp32) and next h (packed bf16).
    unsigned short* hnext = (unsigned short*)Hbuf[(t + 1) & 1];
#pragma unroll
    for (int j = 0; j < 2; ++j) {
      int hcol = (ng * 2 + j) * 16 + nlane;
#pragma unroll
      for (int r = 0; r < 8; ++r) {
        int brow = mt * 16 + klane8 + r;
        float val = tanhf(c[j][r] + xu[j][r]);
        out[((size_t)brow * T + t) * H + hcol] = val;
        hnext[hidx[j][r]] = f2bf(val);
        if (t == T - 1)
          out[(size_t)B * T * H + (size_t)brow * H + hcol] = val;
      }
    }
    grid_barrier(cnt, gen, nblk);   // includes __syncthreads (protects Hl reuse)
  }
}

// ---------------------------------------------------------------------------
// Launch
// ---------------------------------------------------------------------------
extern "C" void kernel_launch(void* const* d_in, const int* in_sizes, int n_in,
                              void* d_out, int out_size, void* d_ws, size_t ws_size,
                              hipStream_t stream) {
  const float* x     = (const float*)d_in[0];  // (64,512,1024)
  const float* state = (const float*)d_in[1];  // (64,1024)
  const float* W     = (const float*)d_in[2];  // (1024,1024)
  const float* U     = (const float*)d_in[3];  // (1024,1024)
  const float* bias  = (const float*)d_in[4];  // (1024,)
  float* out = (float*)d_out;

  char* ws = (char*)d_ws;
  unsigned int* Wp  = (unsigned int*)(ws);                             // 2 MB
  unsigned int* Up  = (unsigned int*)(ws + (2u << 20));                // 2 MB
  unsigned int* Hp0 = (unsigned int*)(ws + (4u << 20));                // 128 KB
  unsigned int* Hp1 = (unsigned int*)(ws + (4u << 20) + (128u << 10)); // 128 KB
  int* barc = (int*)(ws + (4u << 20) + (256u << 10));
  int* barg = barc + 1;
  unsigned int* Xp  = (unsigned int*)(ws + (5u << 20));                // 64 MB

  const size_t WS_NEEDED_PACKED = (5u << 20) + ((size_t)64 << 20);

  pack_bfrag<<<4096, 256, 0, stream>>>(W, (unsigned short*)Wp);
  pack_bfrag<<<4096, 256, 0, stream>>>(U, (unsigned short*)Up);
  pack_afrag<<<128, 256, 0, stream>>>(state, Hp0);   // 32768 u32 pairs
  init_barrier<<<1, 32, 0, stream>>>(barc, barg);

  if (ws_size >= WS_NEEDED_PACKED) {
    pack_afrag<<<65536, 256, 0, stream>>>(x, Xp);
    xu_gemm_packed<<<4096, 256, 0, stream>>>(Xp, Up, bias, out);
  } else {
    xu_gemm_convert<<<4096, 256, 0, stream>>>(x, Up, bias, out);
  }

  rnn_scan<<<16, 256, 0, stream>>>(Wp, out, Hp0, Hp1, barc, barg);

  (void)in_sizes; (void)n_in; (void)out_size;
}
